// cndtn_dcgan_lstm_47536698032568
// MI455X (gfx1250) — matrix operand order repaired and checked by reading, not compiled, not run on hardware
//
#include <hip/hip_runtime.h>
#include <hip/hip_bf16.h>
#include <math.h>
#include <stdint.h>

typedef __attribute__((ext_vector_type(16))) __bf16 v16bf;
typedef __attribute__((ext_vector_type(8)))  float  v8f;

static __device__ __forceinline__ float sigmoidf_(float x) {
  return 1.0f / (1.0f + __expf(-x));
}

// Async copy 16 bytes global -> LDS (GLOBAL_LOAD_ASYNC_TO_LDS_B128, ASYNCcnt).
static __device__ __forceinline__ void async_b128(uint32_t lds_byte,
                                                  const float* gaddr) {
  asm volatile("global_load_async_to_lds_b128 %0, %1, off"
               :: "v"(lds_byte), "v"(gaddr)
               : "memory");
}
static __device__ __forceinline__ void wait_async0() {
  asm volatile("s_wait_asynccnt 0x0" ::: "memory");
}

// ---------------------------------------------------------------------------
__global__ __launch_bounds__(256) void zero_f32(float* __restrict__ p, int n) {
  int i = blockIdx.x * 256 + threadIdx.x;
  if (i < n) p[i] = 0.0f;
}

__global__ __launch_bounds__(256) void add_vec(const float* __restrict__ a,
                                               const float* __restrict__ b,
                                               float* __restrict__ o, int n) {
  int i = blockIdx.x * 256 + threadIdx.x;
  if (i < n) o[i] = a[i] + b[i];
}

// ---------------------------------------------------------------------------
// WMMA bf16 GEMM with double-buffered async-to-LDS staging.
//   C[N,M] = act( A[N,K] @ W[M,K]^T + bias [+ addSrc] )
//   Block tile 64x256; 8 waves as 2(row) x 4(col); wave tile 32x64 = 8 WMMA.
//   Per k-step: A tile 64x32 fp32 (8KB) + W tile 256x32 fp32 (32KB) staged in
//   LDS via GLOBAL_LOAD_ASYNC_TO_LDS_B128.
//   Rows padded to 36 floats: 144B row stride keeps every 16B chunk aligned
//   (ds_load_b128) and r*36 mod 64 hits 16 distinct banks -> conflict-free.
// ---------------------------------------------------------------------------
#define APAD 36
#define A_TILE (64 * APAD)
#define W_TILE (256 * APAD)

__global__ __launch_bounds__(256) void gemm_wmma_bf16(
    const float* __restrict__ A, const float* __restrict__ W,
    const float* __restrict__ bias, float* __restrict__ C,
    const float* __restrict__ addSrc, int addRowMul, int addRowOff,
    int N, int M, int K, int relu) {
  alignas(16) __shared__ float lds[2][A_TILE + W_TILE];

  const int tid  = threadIdx.x;
  const int lane = tid & 31;
  const int wave = tid >> 5;
  const int r    = lane & 15;   // A row / D column within a 16x16 tile
  const int hi   = lane >> 4;   // half-wave selector
  const int wrow = wave >> 2;   // 0..1  (32-row strip)
  const int wcol = wave & 3;    // 0..3  (64-col strip)

  const int i0blk = blockIdx.x * 64;
  const int j0blk = blockIdx.y * 256;

  const uint32_t ldsBase = (uint32_t)(uintptr_t)&lds[0][0];

  // Issue async copies of the k0 tile into buffer `buf`.
  auto stage = [&](int buf, int k0) {
    const uint32_t base = ldsBase + (uint32_t)buf * (A_TILE + W_TILE) * 4u;
    // A tile: 64 rows x 8 chunks(16B) = 512 transfers, 2 per thread.
#pragma unroll
    for (int it = 0; it < 2; ++it) {
      const int id  = tid + 256 * it;
      const int row = id >> 3, ch = id & 7;
      const float* g = A + (size_t)(i0blk + row) * K + k0 + ch * 4;
      async_b128(base + (uint32_t)(row * APAD + ch * 4) * 4u, g);
    }
    // W tile: 256 rows x 8 chunks = 2048 transfers, 8 per thread.
#pragma unroll
    for (int it = 0; it < 8; ++it) {
      const int id  = tid + 256 * it;
      const int row = id >> 3, ch = id & 7;
      const float* g = W + (size_t)(j0blk + row) * K + k0 + ch * 4;
      if (it == 0) __builtin_prefetch(g + 32, 0, 1);  // next K tile into GL2
      async_b128(base + (uint32_t)(A_TILE + row * APAD + ch * 4) * 4u, g);
    }
  };

  v8f acc[2][4];
#pragma unroll
  for (int it = 0; it < 2; ++it)
#pragma unroll
    for (int jt = 0; jt < 4; ++jt)
#pragma unroll
      for (int e = 0; e < 8; ++e) acc[it][jt][e] = 0.0f;

  stage(0, 0);
  int buf = 0;
  for (int k0 = 0; k0 < K; k0 += 32) {
    wait_async0();      // this wave's share of buf has landed
    __syncthreads();    // everyone's share has landed; prior reads done
    if (k0 + 32 < K) stage(buf ^ 1, k0 + 32);

    const float* As = &lds[buf][0];
    const float* Ws = &lds[buf][A_TILE];

    // A fragments (16-bit A 16x32 layout): lane holds row r,
    //   elems 0..7 -> K = 8*hi + e ; elems 8..15 -> K = 16 + 8*hi + e.
    v16bf afr[2];
#pragma unroll
    for (int it = 0; it < 2; ++it) {
      const float* ar = As + (wrow * 32 + it * 16 + r) * APAD;
      const float4 a0 = *(const float4*)(ar + 8 * hi);
      const float4 a1 = *(const float4*)(ar + 8 * hi + 4);
      const float4 a2 = *(const float4*)(ar + 16 + 8 * hi);
      const float4 a3 = *(const float4*)(ar + 16 + 8 * hi + 4);
      afr[it][0]  = (__bf16)a0.x; afr[it][1]  = (__bf16)a0.y;
      afr[it][2]  = (__bf16)a0.z; afr[it][3]  = (__bf16)a0.w;
      afr[it][4]  = (__bf16)a1.x; afr[it][5]  = (__bf16)a1.y;
      afr[it][6]  = (__bf16)a1.z; afr[it][7]  = (__bf16)a1.w;
      afr[it][8]  = (__bf16)a2.x; afr[it][9]  = (__bf16)a2.y;
      afr[it][10] = (__bf16)a2.z; afr[it][11] = (__bf16)a2.w;
      afr[it][12] = (__bf16)a3.x; afr[it][13] = (__bf16)a3.y;
      afr[it][14] = (__bf16)a3.z; afr[it][15] = (__bf16)a3.w;
    }

#pragma unroll
    for (int jt = 0; jt < 4; ++jt) {
      // B fragment (32x16): lane holds column n = r, elems 0..7 -> K = 8*hi + e,
      // elems 8..15 -> K = 16 + 8*hi + e.  B(k,n) = W[n,k].
      const float* wr = Ws + (wcol * 64 + jt * 16 + r) * APAD + 8 * hi;
      const float4 b0 = *(const float4*)(wr);
      const float4 b1 = *(const float4*)(wr + 4);
      const float4 b2 = *(const float4*)(wr + 16);
      const float4 b3 = *(const float4*)(wr + 20);
      v16bf bw;
      bw[0]  = (__bf16)b0.x; bw[1]  = (__bf16)b0.y;
      bw[2]  = (__bf16)b0.z; bw[3]  = (__bf16)b0.w;
      bw[4]  = (__bf16)b1.x; bw[5]  = (__bf16)b1.y;
      bw[6]  = (__bf16)b1.z; bw[7]  = (__bf16)b1.w;
      bw[8]  = (__bf16)b2.x; bw[9]  = (__bf16)b2.y;
      bw[10] = (__bf16)b2.z; bw[11] = (__bf16)b2.w;
      bw[12] = (__bf16)b3.x; bw[13] = (__bf16)b3.y;
      bw[14] = (__bf16)b3.z; bw[15] = (__bf16)b3.w;
#pragma unroll
      for (int it = 0; it < 2; ++it)
        acc[it][jt] = __builtin_amdgcn_wmma_f32_16x16x32_bf16(
            false, afr[it], false, bw, (short)0, acc[it][jt], false, false);
    }
    buf ^= 1;
  }

  // Epilogue. D layout: lane holds column n = r, rows v + 8*hi across 8 VGPRs.
#pragma unroll
  for (int it = 0; it < 2; ++it) {
#pragma unroll
    for (int jt = 0; jt < 4; ++jt) {
      const int j  = j0blk + wcol * 64 + jt * 16 + r;
      const float bv = bias ? bias[j] : 0.0f;
#pragma unroll
      for (int v = 0; v < 8; ++v) {
        const int row = i0blk + wrow * 32 + it * 16 + v + 8 * hi;
        float cval = acc[it][jt][v] + bv;
        if (addSrc)
          cval += addSrc[((size_t)row * addRowMul + addRowOff) * M + j];
        if (relu) cval = fmaxf(cval, 0.0f);
        C[(size_t)row * M + j] = cval;
      }
    }
  }
}

// ---------------------------------------------------------------------------
// xfdot[n] = sum_k xf[n,k] * fcW[1024+k]
// ---------------------------------------------------------------------------
__global__ __launch_bounds__(256) void xf_dot(const float* __restrict__ xf,
                                              const float* __restrict__ fcW,
                                              float* __restrict__ xfdot) {
  __shared__ float red[256];
  const int n = blockIdx.x;
  float p = 0.0f;
  for (int k = threadIdx.x; k < 1024; k += 256)
    p += xf[(size_t)n * 1024 + k] * fcW[1024 + k];
  red[threadIdx.x] = p;
  __syncthreads();
  for (int s = 128; s > 0; s >>= 1) {
    if (threadIdx.x < s) red[threadIdx.x] += red[threadIdx.x + s];
    __syncthreads();
  }
  if (threadIdx.x == 0) xfdot[n] = red[0];
}

// ---------------------------------------------------------------------------
// LSTM pointwise + FC head. One block per batch row b.
// ---------------------------------------------------------------------------
__global__ __launch_bounds__(256) void lstm_pointwise(
    const float* __restrict__ gates, float* __restrict__ h,
    float* __restrict__ c, const float* __restrict__ fcW,
    const float* __restrict__ xfdot, const float* __restrict__ fcb,
    float* __restrict__ out, int t) {
  __shared__ float red[256];
  const int b = blockIdx.x;
  const float* g = gates + (size_t)b * 4096;
  float partial = 0.0f;
  for (int k = threadIdx.x; k < 1024; k += 256) {
    const float ig = sigmoidf_(g[k]);
    const float fg = sigmoidf_(g[1024 + k]);
    const float gg = tanhf(g[2048 + k]);
    const float og = sigmoidf_(g[3072 + k]);
    const float cn = fg * c[b * 1024 + k] + ig * gg;
    c[b * 1024 + k] = cn;
    const float hn = og * tanhf(cn);
    h[b * 1024 + k] = hn;
    partial += hn * fcW[k];
  }
  red[threadIdx.x] = partial;
  __syncthreads();
  for (int s = 128; s > 0; s >>= 1) {
    if (threadIdx.x < s) red[threadIdx.x] += red[threadIdx.x + s];
    __syncthreads();
  }
  if (threadIdx.x == 0)
    out[t * 64 + b] = sigmoidf_(red[0] + xfdot[b * 20 + t] + fcb[0]);
}

// ---------------------------------------------------------------------------
extern "C" void kernel_launch(void* const* d_in, const int* in_sizes, int n_in,
                              void* d_out, int out_size, void* d_ws,
                              size_t ws_size, hipStream_t stream) {
  (void)in_sizes; (void)n_in; (void)out_size; (void)ws_size;
  const float* input_frames = (const float*)d_in[0];   // [64,20,12288]
  const float* next_frames  = (const float*)d_in[1];
  const float* encR_W1 = (const float*)d_in[2];
  const float* encR_b1 = (const float*)d_in[3];
  const float* encR_W2 = (const float*)d_in[4];
  const float* encR_b2 = (const float*)d_in[5];
  const float* encR_W3 = (const float*)d_in[6];
  const float* encR_b3 = (const float*)d_in[7];
  const float* encF_W1 = (const float*)d_in[8];
  const float* encF_b1 = (const float*)d_in[9];
  const float* encF_W2 = (const float*)d_in[10];
  const float* encF_b2 = (const float*)d_in[11];
  const float* encF_W3 = (const float*)d_in[12];
  const float* encF_b3 = (const float*)d_in[13];
  const float* W_ih = (const float*)d_in[14];          // [4096,1024]
  const float* W_hh = (const float*)d_in[15];          // [4096,1024]
  const float* b_ih = (const float*)d_in[16];
  const float* b_hh = (const float*)d_in[17];
  const float* fc_W = (const float*)d_in[18];          // [1,2048]
  const float* fc_b = (const float*)d_in[19];
  float* out = (float*)d_out;                          // [20,64]

  // Workspace layout (floats).
  float* ws = (float*)d_ws;
  size_t o = 0;
  float* act1  = ws + o; o += 1280ull * 4096;
  float* act2  = ws + o; o += 1280ull * 2048;
  float* xr    = ws + o; o += 1280ull * 1024;
  float* xf    = ws + o; o += 1280ull * 1024;
  float* gx    = ws + o; o += 1280ull * 4096;
  float* gates = ws + o; o += 64ull * 4096;
  float* hbuf  = ws + o; o += 64ull * 1024;
  float* cbuf  = ws + o; o += 64ull * 1024;   // contiguous with h
  float* xfd   = ws + o; o += 1280;
  float* bsum  = ws + o; o += 4096;

  const dim3 blk(256);

  zero_f32<<<(64 * 1024 * 2 + 255) / 256, blk, 0, stream>>>(hbuf, 64 * 1024 * 2);
  add_vec<<<(4096 + 255) / 256, blk, 0, stream>>>(b_ih, b_hh, bsum, 4096);

  auto gemm = [&](const float* A, const float* W, const float* b, float* Cp,
                  int N, int M, int K, int relu) {
    dim3 grid(N / 64, M / 256);
    gemm_wmma_bf16<<<grid, blk, 0, stream>>>(A, W, b, Cp, nullptr, 0, 0, N, M,
                                             K, relu);
  };

  // Encoders over all B*T = 1280 frames at once.
  gemm(input_frames, encR_W1, encR_b1, act1, 1280, 4096, 12288, 1);
  gemm(act1,         encR_W2, encR_b2, act2, 1280, 2048, 4096, 1);
  gemm(act2,         encR_W3, encR_b3, xr,   1280, 1024, 2048, 1);
  gemm(next_frames,  encF_W1, encF_b1, act1, 1280, 4096, 12288, 1);
  gemm(act1,         encF_W2, encF_b2, act2, 1280, 2048, 4096, 1);
  gemm(act2,         encF_W3, encF_b3, xf,   1280, 1024, 2048, 1);

  // Hoisted x-path of the LSTM: gx = xr @ W_ih^T + (b_ih + b_hh).
  gemm(xr, W_ih, bsum, gx, 1280, 4096, 1024, 0);

  // FC-head xf contribution for all (b,t).
  xf_dot<<<1280, blk, 0, stream>>>(xf, fc_W, xfd);

  // Sequential LSTM: per step only h @ W_hh^T (W_hh is L2-resident).
  for (int t = 0; t < 20; ++t) {
    dim3 gridh(1, 16);  // N=64 rows, M=4096 cols
    gemm_wmma_bf16<<<gridh, blk, 0, stream>>>(hbuf, W_hh, nullptr, gates, gx,
                                              20, t, 64, 4096, 1024, 0);
    lstm_pointwise<<<64, blk, 0, stream>>>(gates, hbuf, cbuf, fc_W, xfd, fc_b,
                                           out, t);
  }
}
